// Net_56770877718819
// MI455X (gfx1250) — matrix operand order repaired and checked by reading, not compiled, not run on hardware
//
#include <hip/hip_runtime.h>
#include <hip/hip_bf16.h>
#include <hip/hip_fp16.h>

typedef __attribute__((ext_vector_type(16))) _Float16 v16h;
typedef __attribute__((ext_vector_type(8)))  float    v8f;
typedef __attribute__((ext_vector_type(4)))  unsigned u32x4;

#define N_NODE 228
#define T_IN   12
#define S_LEN  2736            // N_NODE * T_IN
#define G_F    64
#define HEADS  2
#define HD     32
#define DFF    2048
#define FCO    256
#define BATCH  8
#define ROWS   (BATCH * S_LEN) // 21888
#define SP     2752            // S padded to multiple of 32 for attention K/V
#define EPS    1e-5f

union frag16 { v16h v; u32x4 q[2]; unsigned u[8]; _Float16 h[16]; };

__device__ __forceinline__ v8f wmma_f16(v16h a, v16h b, v8f c) {
  return __builtin_amdgcn_wmma_f32_16x16x32_f16(false, a, false, b, (short)0, c, false, false);
}

// A fragment: 16x32 f16 row-major tile at (row0,k0), ld in elements.
// Layout (ISA 7.12.2): lane m=lane&15, hs=lane>>4; halves 0..7 = k[hs*8 .. hs*8+7],
// halves 8..15 = k[16+hs*8 .. 16+hs*8+7]  ->  two contiguous 128-bit loads.
__device__ __forceinline__ v16h load_a_frag(const _Float16* base, int ld, int lane) {
  frag16 f;
  int m = lane & 15, hs = lane >> 4;
  const _Float16* p = base + (size_t)m * ld + hs * 8;
  f.q[0] = *(const u32x4*)(p);
  f.q[1] = *(const u32x4*)(p + 16);
  return f.v;
}

// B fragment: 32x16 f16 where B[k][n] = base[n*ld + k] (source holds the 16
// output columns as rows of length >= 32). Halves 0..7 = k[hs*8 .. hs*8+7],
// halves 8..15 = k[16+hs*8 .. 16+hs*8+7] -> two contiguous 128-bit loads.
__device__ __forceinline__ v16h load_b_frag(const _Float16* base, int ld, int lane) {
  frag16 f;
  int n = lane & 15, hs = lane >> 4;
  const _Float16* p = base + (size_t)n * ld + hs * 8;
  f.q[0] = *(const u32x4*)(p);
  f.q[1] = *(const u32x4*)(p + 16);
  return f.v;
}

__device__ __forceinline__ float wave_reduce_add(float v) {
#pragma unroll
  for (int m = 16; m >= 1; m >>= 1) v += __shfl_xor(v, m, 32);
  return v;
}
__device__ __forceinline__ float half_reduce_add(float v) {
#pragma unroll
  for (int m = 8; m >= 1; m >>= 1) v += __shfl_xor(v, m, 16);
  return v;
}
__device__ __forceinline__ float half_reduce_max(float v) {
#pragma unroll
  for (int m = 8; m >= 1; m >>= 1) v = fmaxf(v, __shfl_xor(v, m, 16));
  return v;
}

// ---------------- weight f32 -> f16 cast ----------------
__global__ void k_cvt(const float* __restrict__ src, _Float16* __restrict__ dst, int n) {
  int i = blockIdx.x * 256 + threadIdx.x;
  if (i < n) dst[i] = (_Float16)src[i];
}

// ---------------- K0: adaptive GCN + BN1 + ReLU + residual ----------------
// grid = S_LEN blocks, 64 threads (thread = output feature g)
__global__ void k_gcn(const float* __restrict__ x, const float* __restrict__ ne,
                      const float* __restrict__ gcw, const float* __restrict__ gcb,
                      const float* __restrict__ bn1g, const float* __restrict__ bn1b,
                      float* __restrict__ h32, _Float16* __restrict__ h16) {
  int s = blockIdx.x, g = threadIdx.x;
  float gw = 0.f, gb = 0.f;
#pragma unroll
  for (int e = 0; e < 16; ++e) {
    float w = ne[s * 16 + e];
    gw += w * gcw[e * 64 + g];
    gb += w * gcb[e * 64 + g];
  }
  float xr[8], x2[8], sum = 0.f, sq = 0.f;
#pragma unroll
  for (int b = 0; b < 8; ++b) {
    float xv = x[(size_t)b * S_LEN + s];
    float v = xv * gw + gb;
    xr[b] = xv; x2[b] = v; sum += v; sq += v * v;
  }
  sum = wave_reduce_add(sum); sq = wave_reduce_add(sq);
  __shared__ float rs[2], rq[2];
  int wid = threadIdx.x >> 5;
  if ((threadIdx.x & 31) == 0) { rs[wid] = sum; rq[wid] = sq; }
  __syncthreads();
  float ts = rs[0] + rs[1], tq = rq[0] + rq[1];
  float mean = ts * (1.f / 512.f);
  float var = tq * (1.f / 512.f) - mean * mean;
  float inv = rsqrtf(var + EPS);
  float gg = bn1g[s], bb = bn1b[s];
#pragma unroll
  for (int b = 0; b < 8; ++b) {
    float bnv = (x2[b] - mean) * inv * gg + bb;
    float hv = (bnv > 0.f ? bnv : 0.f) + xr[b];
    size_t idx = ((size_t)b * S_LEN + s) * 64 + g;
    h32[idx] = hv; h16[idx] = (_Float16)hv;
  }
}

// ---------------- K1: fused QKV projection (WMMA) ----------------
// jobs = 1368 mtiles * 12 (3 mats x 4 ntiles); 8 waves/block -> 2052 blocks
__global__ void k_qkv(const _Float16* __restrict__ h16,
                      const _Float16* __restrict__ wq, const _Float16* __restrict__ wk,
                      const _Float16* __restrict__ wv,
                      const float* __restrict__ bq, const float* __restrict__ bk,
                      const float* __restrict__ bv,
                      _Float16* __restrict__ qb, _Float16* __restrict__ kb,
                      _Float16* __restrict__ vtb) {
  int wid = threadIdx.x >> 5, lane = threadIdx.x & 31;
  int job = blockIdx.x * 8 + wid;
  int mt = job / 12, c = job % 12;
  int mat = c >> 2, nt = c & 3;
  const _Float16* W = (mat == 0) ? wq : ((mat == 1) ? wk : wv);
  const float* bias = (mat == 0) ? bq : ((mat == 1) ? bk : bv);
  v8f acc = {0.f,0.f,0.f,0.f,0.f,0.f,0.f,0.f};
  const _Float16* Ab = h16 + (size_t)mt * 16 * 64;
#pragma unroll
  for (int kk = 0; kk < 2; ++kk) {
    v16h a = load_a_frag(Ab + kk * 32, 64, lane);
    v16h bf = load_b_frag(W + (size_t)nt * 16 * 64 + kk * 32, 64, lane);
    acc = wmma_f16(a, bf, acc);
  }
  int n = lane & 15, hs = lane >> 4;
  int col = nt * 16 + n;
  int head = col >> 5, d = col & 31;
  float bvl = bias[col];
#pragma unroll
  for (int r = 0; r < 8; ++r) {
    int row = mt * 16 + r + 8 * hs;
    int bb = row / S_LEN, s = row - bb * S_LEN;
    _Float16 hv = (_Float16)(acc[r] + bvl);
    if (mat == 0)      qb[((size_t)(bb * HEADS + head) * S_LEN + s) * HD + d] = hv;
    else if (mat == 1) kb[((size_t)(bb * HEADS + head) * SP + s) * HD + d] = hv;
    else               vtb[((size_t)(bb * HEADS + head) * HD + d) * SP + s] = hv;
  }
}

// ---------------- K2: block-causal flash attention ----------------
// grid = (ceil(171/4), B*H), block = 128 (1 wave = one 16-row query tile)
__global__ void k_attn(const _Float16* __restrict__ qb, const _Float16* __restrict__ kb,
                       const _Float16* __restrict__ vtb, _Float16* __restrict__ attn) {
  __shared__ alignas(16) _Float16 Pb[4][16][32];
  int wid = threadIdx.x >> 5, lane = threadIdx.x & 31;
  int qt = blockIdx.x * 4 + wid;
  if (qt >= S_LEN / 16) return;
  int bh = blockIdx.y;
  int b = bh >> 1, head = bh & 1;
  const _Float16* Qp = qb + (size_t)bh * S_LEN * HD;
  const _Float16* Kp = kb + (size_t)bh * SP * HD;
  const _Float16* Vt = vtb + (size_t)bh * HD * SP;
  int q0 = qt * 16;
  int n = lane & 15, hs = lane >> 4;
  v16h aQ = load_a_frag(Qp + (size_t)q0 * HD, HD, lane);
  int lim[8];
#pragma unroll
  for (int r = 0; r < 8; ++r) { int m = q0 + r + 8 * hs; lim[r] = (m / N_NODE + 1) * N_NODE; }
  int limMax = ((q0 + 15) / N_NODE + 1) * N_NODE;
  float mS[8], lS[8];
#pragma unroll
  for (int r = 0; r < 8; ++r) { mS[r] = -1e30f; lS[r] = 0.f; }
  v8f acc0 = {0.f,0.f,0.f,0.f,0.f,0.f,0.f,0.f};
  v8f acc1 = {0.f,0.f,0.f,0.f,0.f,0.f,0.f,0.f};
  v8f zero = {0.f,0.f,0.f,0.f,0.f,0.f,0.f,0.f};
  const float scale = 0.17677669529663689f;   // 1/sqrt(32)
  for (int kp = 0; kp < limMax; kp += 32) {
    v16h bk0 = load_b_frag(Kp + (size_t)kp * HD, HD, lane);
    v16h bk1 = load_b_frag(Kp + (size_t)(kp + 16) * HD, HD, lane);
    v8f s0 = wmma_f16(aQ, bk0, zero);
    v8f s1 = wmma_f16(aQ, bk1, zero);
#pragma unroll
    for (int r = 0; r < 8; ++r) {
      float a0 = (kp + n      < lim[r]) ? s0[r] * scale : -1e30f;
      float a1 = (kp + 16 + n < lim[r]) ? s1[r] * scale : -1e30f;
      float tm = half_reduce_max(fmaxf(a0, a1));
      float nm = fmaxf(mS[r], tm);
      float alpha = __expf(mS[r] - nm);
      mS[r] = nm;
      float e0 = __expf(a0 - nm), e1 = __expf(a1 - nm);
      float rsum = half_reduce_add(e0 + e1);
      lS[r] = lS[r] * alpha + rsum;
      acc0[r] *= alpha; acc1[r] *= alpha;
      Pb[wid][r + 8 * hs][n] = (_Float16)e0;
      Pb[wid][r + 8 * hs][n + 16] = (_Float16)e1;
    }
    v16h aP = load_a_frag(&Pb[wid][0][0], 32, lane);
    v16h bv0 = load_b_frag(Vt + kp, SP, lane);
    v16h bv1 = load_b_frag(Vt + (size_t)16 * SP + kp, SP, lane);
    acc0 = wmma_f16(aP, bv0, acc0);
    acc1 = wmma_f16(aP, bv1, acc1);
  }
#pragma unroll
  for (int r = 0; r < 8; ++r) {
    float invl = 1.f / lS[r];
    int s = q0 + r + 8 * hs;
    size_t ro = ((size_t)b * S_LEN + s) * 64 + head * 32;
    attn[ro + n] = (_Float16)(acc0[r] * invl);
    attn[ro + 16 + n] = (_Float16)(acc1[r] * invl);
  }
}

// ---------------- K3: output projection + residual ----------------
// jobs = 1368*4, 8 waves/block -> 684 blocks
__global__ void k_oproj(const _Float16* __restrict__ a16, const _Float16* __restrict__ wo,
                        const float* __restrict__ bo, const float* __restrict__ hres,
                        float* __restrict__ out) {
  int wid = threadIdx.x >> 5, lane = threadIdx.x & 31;
  int job = blockIdx.x * 8 + wid;
  int mt = job >> 2, nt = job & 3;
  v8f acc = {0.f,0.f,0.f,0.f,0.f,0.f,0.f,0.f};
#pragma unroll
  for (int kk = 0; kk < 2; ++kk) {
    v16h a = load_a_frag(a16 + (size_t)mt * 16 * 64 + kk * 32, 64, lane);
    v16h bf = load_b_frag(wo + (size_t)nt * 16 * 64 + kk * 32, 64, lane);
    acc = wmma_f16(a, bf, acc);
  }
  int n = lane & 15, hs = lane >> 4;
  int col = nt * 16 + n;
  float bb = bo[col];
#pragma unroll
  for (int r = 0; r < 8; ++r) {
    int row = mt * 16 + r + 8 * hs;
    out[(size_t)row * 64 + col] = acc[r] + bb + hres[(size_t)row * 64 + col];
  }
}

// ---------------- LayerNorm over G=64 (fp32 in, fp32+f16 out) ----------------
// grid = ROWS/8, 256 threads (wave per row)
__global__ void k_ln(const float* __restrict__ src, const float* __restrict__ g,
                     const float* __restrict__ b, float* __restrict__ d32,
                     _Float16* __restrict__ d16) {
  int wid = threadIdx.x >> 5, lane = threadIdx.x & 31;
  int row = blockIdx.x * 8 + wid;
  const float* p = src + (size_t)row * 64;
  float v0 = p[lane], v1 = p[lane + 32];
  float mean = wave_reduce_add(v0 + v1) * (1.f / 64.f);
  float d0 = v0 - mean, d1 = v1 - mean;
  float inv = rsqrtf(wave_reduce_add(d0 * d0 + d1 * d1) * (1.f / 64.f) + EPS);
  float o0 = g[lane] * d0 * inv + b[lane];
  float o1 = g[lane + 32] * d1 * inv + b[lane + 32];
  size_t base = (size_t)row * 64;
  d32[base + lane] = o0; d32[base + lane + 32] = o1;
  d16[base + lane] = (_Float16)o0; d16[base + lane + 32] = (_Float16)o1;
}

// ---------------- K4/K5 fused FFN: relu(h1@W1.T+b1)@W2.T + b2 + h1 ----------------
// grid = 1368, block = 256. FFN1 chunks of 512 staged in LDS; FFN2 accumulates.
__global__ void k_ffn(const _Float16* __restrict__ h16, const float* __restrict__ hres,
                      const _Float16* __restrict__ w1, const float* __restrict__ b1,
                      const _Float16* __restrict__ w2, const float* __restrict__ b2,
                      float* __restrict__ out) {
  __shared__ alignas(16) _Float16 cA[16][512];
  int wid = threadIdx.x >> 5, lane = threadIdx.x & 31;
  int n = lane & 15, hs = lane >> 4;
  int rowbase = blockIdx.x * 16;
  // A tile (16x64) is invariant across the whole kernel: load its 2 fragments once.
  v16h af0 = load_a_frag(h16 + (size_t)rowbase * 64, 64, lane);
  v16h af1 = load_a_frag(h16 + (size_t)rowbase * 64 + 32, 64, lane);
  v8f acc2 = {0.f,0.f,0.f,0.f,0.f,0.f,0.f,0.f};
  for (int ch = 0; ch < 4; ++ch) {
#pragma unroll
    for (int i = 0; i < 4; ++i) {
      int ltile = wid * 4 + i;
      int ncol0 = ch * 512 + ltile * 16;
      v8f a1 = wmma_f16(af0, load_b_frag(w1 + (size_t)ncol0 * 64, 64, lane),
                        (v8f){0.f,0.f,0.f,0.f,0.f,0.f,0.f,0.f});
      a1 = wmma_f16(af1, load_b_frag(w1 + (size_t)ncol0 * 64 + 32, 64, lane), a1);
      float bb = b1[ncol0 + n];
      int lc = ltile * 16 + n;
#pragma unroll
      for (int r = 0; r < 8; ++r) {
        float v = a1[r] + bb;
        cA[r + 8 * hs][lc] = (_Float16)(v > 0.f ? v : 0.f);
      }
    }
    __syncthreads();
    if (wid < 4) {
      if (ch < 3)  // gfx1250 global_prefetch of next W2 chunk
        __builtin_prefetch(w2 + (size_t)(wid * 16) * DFF + (ch + 1) * 512, 0, 1);
#pragma unroll
      for (int kk = 0; kk < 16; ++kk) {
        v16h af = load_a_frag(&cA[0][0] + kk * 32, 512, lane);
        v16h bf = load_b_frag(w2 + (size_t)(wid * 16) * DFF + ch * 512 + kk * 32, DFF, lane);
        acc2 = wmma_f16(af, bf, acc2);
      }
    }
    __syncthreads();
  }
  if (wid < 4) {
    int col = wid * 16 + n;
    float bb = b2[col];
#pragma unroll
    for (int r = 0; r < 8; ++r) {
      int row = rowbase + r + 8 * hs;
      out[(size_t)row * 64 + col] = acc2[r] + bb + hres[(size_t)row * 64 + col];
    }
  }
}

// ---------------- K6: head GEMM [B*N,768] @ FC_W.T -> out1 [B,N,256] ----------------
// jobs = 114*16, 8 waves/block -> 228 blocks. A gathered from h2 (t,g)->k mapping;
// each 8-half run stays inside one t-block, so two b128 loads per k-step.
__global__ void k_head(const _Float16* __restrict__ h2, const _Float16* __restrict__ fcw,
                       const float* __restrict__ fcb, float* __restrict__ out1) {
  int wid = threadIdx.x >> 5, lane = threadIdx.x & 31;
  int job = blockIdx.x * 8 + wid;
  int mt = job >> 4, nt = job & 15;
  int m = lane & 15, hs = lane >> 4, n = m;
  int arow = mt * 16 + m;
  int ab = arow / N_NODE, ann = arow - ab * N_NODE;
  v8f acc = {0.f,0.f,0.f,0.f,0.f,0.f,0.f,0.f};
  for (int k0 = 0; k0 < T_IN * 64; k0 += 32) {
    frag16 fa;
    int ka = k0 + hs * 8;
    int kb = k0 + 16 + hs * 8;
    int ta = ka >> 6, ga = ka & 63;
    int tb = kb >> 6, gb = kb & 63;
    fa.q[0] = *(const u32x4*)(h2 + ((size_t)ab * S_LEN + ta * N_NODE + ann) * 64 + ga);
    fa.q[1] = *(const u32x4*)(h2 + ((size_t)ab * S_LEN + tb * N_NODE + ann) * 64 + gb);
    v16h bf = load_b_frag(fcw + (size_t)(nt * 16) * 768 + k0, 768, lane);
    acc = wmma_f16(fa.v, bf, acc);
  }
  int col = nt * 16 + n;
  float bb = fcb[col];
#pragma unroll
  for (int r = 0; r < 8; ++r) {
    int row = mt * 16 + r + 8 * hs;
    out1[(size_t)row * 256 + col] = acc[r] + bb;
  }
}

// ---------------- K7: BN2 stats per node channel ----------------
__global__ void k_bn2stat(const float* __restrict__ out1, float* __restrict__ stats) {
  int n = blockIdx.x, tid = threadIdx.x;
  float s = 0.f, q = 0.f;
#pragma unroll
  for (int b = 0; b < 8; ++b) {
    float v = out1[((size_t)(b * N_NODE + n)) * 256 + tid];
    s += v; q += v * v;
  }
  s = wave_reduce_add(s); q = wave_reduce_add(q);
  __shared__ float rs[8], rq[8];
  if ((tid & 31) == 0) { rs[tid >> 5] = s; rq[tid >> 5] = q; }
  __syncthreads();
  if (tid == 0) {
    float ts = 0.f, tq = 0.f;
#pragma unroll
    for (int i = 0; i < 8; ++i) { ts += rs[i]; tq += rq[i]; }
    float mean = ts * (1.f / 2048.f);
    float var = tq * (1.f / 2048.f) - mean * mean;
    stats[n * 2] = mean; stats[n * 2 + 1] = rsqrtf(var + EPS);
  }
}

// ---------------- K8: BN2 + relu + FC3 -> output ----------------
__global__ void k_final(const float* __restrict__ out1, const float* __restrict__ stats,
                        const float* __restrict__ g2, const float* __restrict__ b2,
                        const float* __restrict__ fc3w, const float* __restrict__ fc3b,
                        float* __restrict__ out) {
  __shared__ float sv[256];
  int bn = blockIdx.x;
  int b = bn / N_NODE, n = bn - b * N_NODE;
  int c = threadIdx.x;
  float mean = stats[n * 2], inv = stats[n * 2 + 1];
  float v = (out1[(size_t)bn * 256 + c] - mean) * inv * g2[n] + b2[n];
  sv[c] = v > 0.f ? v : 0.f;
  __syncthreads();
  if (c < T_IN) {
    float acc = fc3b[c];
    for (int i = 0; i < 256; ++i) acc += sv[i] * fc3w[c * 256 + i];
    out[(size_t)b * S_LEN + c * N_NODE + n] = acc;
  }
}

extern "C" void kernel_launch(void* const* d_in, const int* in_sizes, int n_in,
                              void* d_out, int out_size, void* d_ws, size_t ws_size,
                              hipStream_t stream) {
  const float* x    = (const float*)d_in[0];
  const float* ne   = (const float*)d_in[1];
  const float* gcw  = (const float*)d_in[2];
  const float* gcb  = (const float*)d_in[3];
  const float* bn1g = (const float*)d_in[4];
  const float* bn1b = (const float*)d_in[5];
  const float* bn2g = (const float*)d_in[6];
  const float* bn2b = (const float*)d_in[7];
  const float* Wq = (const float*)d_in[8];  const float* bq = (const float*)d_in[9];
  const float* Wk = (const float*)d_in[10]; const float* bk = (const float*)d_in[11];
  const float* Wv = (const float*)d_in[12]; const float* bvp = (const float*)d_in[13];
  const float* Wo = (const float*)d_in[14]; const float* bo = (const float*)d_in[15];
  const float* ln1g = (const float*)d_in[16]; const float* ln1b = (const float*)d_in[17];
  const float* W1 = (const float*)d_in[18]; const float* b1 = (const float*)d_in[19];
  const float* W2 = (const float*)d_in[20]; const float* b2 = (const float*)d_in[21];
  const float* ln2g = (const float*)d_in[22]; const float* ln2b = (const float*)d_in[23];
  const float* FCW = (const float*)d_in[24]; const float* FCb = (const float*)d_in[25];
  const float* FC3W = (const float*)d_in[26]; const float* FC3b = (const float*)d_in[27];
  float* out = (float*)d_out;

  char* w = (char*)d_ws;
  size_t off = 0;
  auto alloc = [&](size_t bytes) { size_t o = off; off += (bytes + 255) & ~(size_t)255; return o; };

  float*     h32   = (float*)(w + alloc((size_t)ROWS * 64 * 4));
  _Float16*  h16   = (_Float16*)(w + alloc((size_t)ROWS * 64 * 2));
  _Float16*  qb    = (_Float16*)(w + alloc((size_t)BATCH * HEADS * S_LEN * HD * 2));
  _Float16*  kbuf  = (_Float16*)(w + alloc((size_t)BATCH * HEADS * SP * HD * 2));
  _Float16*  vtbuf = (_Float16*)(w + alloc((size_t)BATCH * HEADS * HD * SP * 2));
  _Float16*  attn  = (_Float16*)(w + alloc((size_t)ROWS * 64 * 2));
  float*     tmp   = (float*)(w + alloc((size_t)ROWS * 64 * 4));
  float*     h1_32 = (float*)(w + alloc((size_t)ROWS * 64 * 4));
  _Float16*  h1_16 = (_Float16*)(w + alloc((size_t)ROWS * 64 * 2));
  float*     h2_32 = (float*)(w + alloc((size_t)ROWS * 64 * 4));
  _Float16*  h2_16 = (_Float16*)(w + alloc((size_t)ROWS * 64 * 2));
  float*     out1  = (float*)(w + alloc((size_t)BATCH * N_NODE * 256 * 4));
  float*     stats = (float*)(w + alloc((size_t)N_NODE * 2 * 4));
  _Float16*  wq16  = (_Float16*)(w + alloc(4096 * 2));
  _Float16*  wk16  = (_Float16*)(w + alloc(4096 * 2));
  _Float16*  wv16  = (_Float16*)(w + alloc(4096 * 2));
  _Float16*  wo16  = (_Float16*)(w + alloc(4096 * 2));
  _Float16*  w1_16 = (_Float16*)(w + alloc((size_t)DFF * 64 * 2));
  _Float16*  w2_16 = (_Float16*)(w + alloc((size_t)64 * DFF * 2));
  _Float16*  fcw16 = (_Float16*)(w + alloc((size_t)256 * 768 * 2));

  // weight casts
  k_cvt<<<(4096 + 255) / 256, 256, 0, stream>>>(Wq, wq16, 4096);
  k_cvt<<<(4096 + 255) / 256, 256, 0, stream>>>(Wk, wk16, 4096);
  k_cvt<<<(4096 + 255) / 256, 256, 0, stream>>>(Wv, wv16, 4096);
  k_cvt<<<(4096 + 255) / 256, 256, 0, stream>>>(Wo, wo16, 4096);
  k_cvt<<<(DFF * 64 + 255) / 256, 256, 0, stream>>>(W1, w1_16, DFF * 64);
  k_cvt<<<(64 * DFF + 255) / 256, 256, 0, stream>>>(W2, w2_16, 64 * DFF);
  k_cvt<<<(256 * 768 + 255) / 256, 256, 0, stream>>>(FCW, fcw16, 256 * 768);

  // zero K/V pads (keys [S_LEN, SP))
  hipMemsetAsync(kbuf, 0, (size_t)BATCH * HEADS * SP * HD * 2, stream);
  hipMemsetAsync(vtbuf, 0, (size_t)BATCH * HEADS * HD * SP * 2, stream);

  // pipeline
  k_gcn<<<S_LEN, 64, 0, stream>>>(x, ne, gcw, gcb, bn1g, bn1b, h32, h16);
  k_qkv<<<(ROWS / 16) * 12 / 8, 256, 0, stream>>>(h16, wq16, wk16, wv16, bq, bk, bvp,
                                                  qb, kbuf, vtbuf);
  k_attn<<<dim3((S_LEN / 16 + 3) / 4, BATCH * HEADS), 128, 0, stream>>>(qb, kbuf, vtbuf, attn);
  k_oproj<<<(ROWS / 16) * 4 / 8, 256, 0, stream>>>(attn, wo16, bo, h32, tmp);
  k_ln<<<ROWS / 8, 256, 0, stream>>>(tmp, ln1g, ln1b, h1_32, h1_16);
  k_ffn<<<ROWS / 16, 256, 0, stream>>>(h1_16, h1_32, w1_16, b1, w2_16, b2, tmp);
  k_ln<<<ROWS / 8, 256, 0, stream>>>(tmp, ln2g, ln2b, h2_32, h2_16);
  k_head<<<(BATCH * N_NODE / 16) * 16 / 8, 256, 0, stream>>>(h2_16, fcw16, FCb, out1);
  k_bn2stat<<<N_NODE, 256, 0, stream>>>(out1, stats);
  k_final<<<BATCH * N_NODE, 256, 0, stream>>>(out1, stats, bn2g, bn2b, FC3W, FC3b, out);

  (void)in_sizes; (void)n_in; (void)out_size; (void)ws_size; (void)h2_32;
}
